// NaiveCustomLSTM_2894807957555
// MI455X (gfx1250) — compile-verified, matrix operand order checked by reading, not run on hardware
//
#include <hip/hip_runtime.h>

// Problem constants (match reference).
#define BB 32
#define TT 512
#define DD 512
#define HH 1024
#define KK (HH + DD)   // 1536 : concatenated [h | x] contraction dim

typedef __attribute__((ext_vector_type(16))) __bf16 v16bf;
typedef __attribute__((ext_vector_type(8)))  float  v8f;

static __device__ __forceinline__ v8f wmma_bf16(v16bf a, v16bf b, v8f c) {
    return __builtin_amdgcn_wmma_f32_16x16x32_bf16(false, a, false, b,
                                                   (short)0, c, false, false);
}

// ---------------------------------------------------------------------------
// Prologue 1: convert x (B,T,D) fp32 -> bf16 (same layout).
// ---------------------------------------------------------------------------
__global__ __launch_bounds__(256) void cvt_x_kernel(const float* __restrict__ x,
                                                    __bf16* __restrict__ xb, int n) {
    int i = blockIdx.x * blockDim.x + threadIdx.x;
    if (i < n) xb[i] = (__bf16)x[i];
}

// ---------------------------------------------------------------------------
// Prologue 2: build W[gate][n][k] bf16, k in [0,1536):
//   k <  1024 : V_gate[k][n]        (recurrent weights, transposed)
//   k >= 1024 : U_gate[k-1024][n]   (input weights, transposed)
// Transposed layout => each WMMA B-operand lane reads 32 contiguous bytes.
// ---------------------------------------------------------------------------
__global__ __launch_bounds__(256) void build_w_kernel(
    const float* __restrict__ Vi, const float* __restrict__ Vf,
    const float* __restrict__ Vo, const float* __restrict__ Vg,
    const float* __restrict__ Ui, const float* __restrict__ Uf,
    const float* __restrict__ Uo, const float* __restrict__ Ug,
    __bf16* __restrict__ W) {
    int i = blockIdx.x * blockDim.x + threadIdx.x;          // k fastest -> coalesced stores
    const int n_total = 4 * HH * KK;
    if (i >= n_total) return;
    int g   = i / (HH * KK);
    int rem = i % (HH * KK);
    int n   = rem / KK;
    int k   = rem % KK;
    const float* V = (g == 0) ? Vi : (g == 1) ? Vf : (g == 2) ? Vo : Vg;
    const float* U = (g == 0) ? Ui : (g == 1) ? Uf : (g == 2) ? Uo : Ug;
    float v = (k < HH) ? V[(size_t)k * HH + n] : U[(size_t)(k - HH) * HH + n];
    W[i] = (__bf16)v;
}

// ---------------------------------------------------------------------------
// Prologue 3: init h ping buffer (bf16) from h0 and c buffer (fp32) from c0.
// ---------------------------------------------------------------------------
__global__ __launch_bounds__(256) void init_state_kernel(const float* __restrict__ h0,
                                                         const float* __restrict__ c0,
                                                         __bf16* __restrict__ hb,
                                                         float* __restrict__ cb, int n) {
    int i = blockIdx.x * blockDim.x + threadIdx.x;
    if (i < n) { hb[i] = (__bf16)h0[i]; cb[i] = c0[i]; }
}

// ---------------------------------------------------------------------------
// One LSTM timestep, fully fused:
//   gates(32 x 4096) = [h_{t-1} | x_t] (32x1536, bf16) @ W^T   via WMMA bf16
//   then sigmoid/tanh gate math, c update, h_t out (fp32 to d_out, bf16 ping).
// Grid: 32 workgroups (each owns 32 hidden units j), 256 threads = 8 wave32s.
// Wave w: gate = w/2, N-subtile = w%2 (16 cols), M = 32 via two accumulators.
// Contraction split into two branch-free, unrolled loops (h part, x part) so
// the scheduler can pipeline global_load_b128 clauses under the WMMA chain.
// ---------------------------------------------------------------------------
__global__ __launch_bounds__(256) void lstm_step_kernel(
    const __bf16* __restrict__ W,      // [4][H][K] bf16 (transposed weights)
    const __bf16* __restrict__ xb,     // [B][T][D] bf16
    const __bf16* __restrict__ hprev,  // [B][H] bf16
    __bf16*       __restrict__ hnext,  // [B][H] bf16
    float*        __restrict__ cbuf,   // [B][H] fp32 (in/out)
    const float* __restrict__ bi, const float* __restrict__ bf_,
    const float* __restrict__ bo, const float* __restrict__ bg,
    float* __restrict__ out,           // [B][T][H] fp32
    int t) {
    __shared__ float lds[4][32][32];   // [gate][batch][j_local] pre-activations

    const int tid   = threadIdx.x;
    const int lane  = tid & 31;
    const int wave  = tid >> 5;
    const int g     = wave >> 1;          // gate 0..3 = i,f,o,g
    const int nsub  = wave & 1;           // which 16-col subtile
    const int j0    = blockIdx.x * 32;    // hidden-unit base of this WG
    const int n0    = j0 + nsub * 16;

    // Contiguous-k lane decomposition used here for both A and B (16-bit, K=32):
    //   lanes 0-15: row/col = lane,     K 0..15 ; lanes 16-31: row/col = lane-16, K 16..31
    const int rc    = lane & 15;
    const int khalf = lane >> 4;

    v8f acc0 = {};   // batches 0..15
    v8f acc1 = {};   // batches 16..31

    // B operand base: W[g][n0+rc][k + khalf*16 ...], 32 contiguous bytes per lane.
    const __bf16* __restrict__ bp = W + ((size_t)(g * HH + n0 + rc) * KK) + khalf * 16;

    // ---- Phase 1: contraction over h_{t-1}, K = 0..1023 (branch-free) ----
    {
        const __bf16* __restrict__ a0 = hprev + (size_t)rc        * HH + khalf * 16;
        const __bf16* __restrict__ a1 = hprev + (size_t)(rc + 16) * HH + khalf * 16;
#pragma unroll 4
        for (int k0 = 0; k0 < HH; k0 += 32) {
            v16bf A0 = *(const v16bf*)(a0 + k0);
            v16bf A1 = *(const v16bf*)(a1 + k0);
            v16bf Bt = *(const v16bf*)(bp + k0);
            acc0 = wmma_bf16(A0, Bt, acc0);
            acc1 = wmma_bf16(A1, Bt, acc1);
        }
    }
    // ---- Phase 2: contraction over x_t, K = 1024..1535 (branch-free) ----
    {
        const __bf16* __restrict__ a0 = xb + ((size_t)rc        * TT + t) * DD + khalf * 16;
        const __bf16* __restrict__ a1 = xb + ((size_t)(rc + 16) * TT + t) * DD + khalf * 16;
        const __bf16* __restrict__ b2 = bp + HH;
#pragma unroll 4
        for (int k0 = 0; k0 < DD; k0 += 32) {
            v16bf A0 = *(const v16bf*)(a0 + k0);
            v16bf A1 = *(const v16bf*)(a1 + k0);
            v16bf Bt = *(const v16bf*)(b2 + k0);
            acc0 = wmma_bf16(A0, Bt, acc0);
            acc1 = wmma_bf16(A1, Bt, acc1);
        }
    }

    // C/D layout: VGPR r -> M = r + 8*(lane>=16), N = lane&15. Spill to LDS.
    {
        const int mh = khalf * 8;
        const int nc = nsub * 16 + rc;
#pragma unroll
        for (int r = 0; r < 8; ++r) {
            lds[g][r + mh][nc]      = acc0[r];
            lds[g][r + mh + 16][nc] = acc1[r];
        }
    }
    __syncthreads();

    // Fused gate math: 1024 (b,j) cells, 4 per thread.
#pragma unroll
    for (int e = 0; e < 4; ++e) {
        int idx = e * 256 + tid;
        int b   = idx >> 5;
        int jj  = idx & 31;
        int j   = j0 + jj;
        float pi = lds[0][b][jj] + bi[j];
        float pf = lds[1][b][jj] + bf_[j];
        float po = lds[2][b][jj] + bo[j];
        float pg = lds[3][b][jj] + bg[j];
        float it = 1.0f / (1.0f + __expf(-pi));
        float ft = 1.0f / (1.0f + __expf(-pf));
        float ot = 1.0f / (1.0f + __expf(-po));
        float gt = 2.0f / (1.0f + __expf(-2.0f * pg)) - 1.0f;   // tanh
        size_t cix = (size_t)b * HH + j;
        float cn = ft * cbuf[cix] + it * gt;
        cbuf[cix] = cn;
        float h = ot * (2.0f / (1.0f + __expf(-2.0f * cn)) - 1.0f);
        out[((size_t)b * TT + t) * HH + j] = h;
        hnext[cix] = (__bf16)h;
    }
}

// ---------------------------------------------------------------------------
// Host-side launcher.
// ---------------------------------------------------------------------------
extern "C" void kernel_launch(void* const* d_in, const int* in_sizes, int n_in,
                              void* d_out, int out_size, void* d_ws, size_t ws_size,
                              hipStream_t stream) {
    const float* x  = (const float*)d_in[0];
    const float* h0 = (const float*)d_in[1];
    const float* c0 = (const float*)d_in[2];
    const float* Ui = (const float*)d_in[3];
    const float* Vi = (const float*)d_in[4];
    const float* bi = (const float*)d_in[5];
    const float* Uf = (const float*)d_in[6];
    const float* Vf = (const float*)d_in[7];
    const float* bf_ = (const float*)d_in[8];
    const float* Uo = (const float*)d_in[9];
    const float* Vo = (const float*)d_in[10];
    const float* bo = (const float*)d_in[11];
    const float* Ug = (const float*)d_in[12];
    const float* Vg = (const float*)d_in[13];
    const float* bg = (const float*)d_in[14];
    float* out = (float*)d_out;

    // Workspace layout (~30 MB total; regions 256B aligned).
    char* ws = (char*)d_ws;
    size_t off = 0;
    __bf16* xb = (__bf16*)(ws + off); off += (size_t)BB * TT * DD * sizeof(__bf16);      // 16 MB
    off = (off + 255) & ~(size_t)255;
    __bf16* W  = (__bf16*)(ws + off); off += (size_t)4 * HH * KK * sizeof(__bf16);       // 12 MB
    off = (off + 255) & ~(size_t)255;
    __bf16* hb0 = (__bf16*)(ws + off); off += (size_t)BB * HH * sizeof(__bf16);          // 64 KB
    off = (off + 255) & ~(size_t)255;
    __bf16* hb1 = (__bf16*)(ws + off); off += (size_t)BB * HH * sizeof(__bf16);          // 64 KB
    off = (off + 255) & ~(size_t)255;
    float* cbuf = (float*)(ws + off); off += (size_t)BB * HH * sizeof(float);            // 128 KB

    // Prologue (re-run every call: ws is not persistent across calls).
    {
        int n = BB * TT * DD;
        cvt_x_kernel<<<(n + 255) / 256, 256, 0, stream>>>(x, xb, n);
    }
    {
        int n = 4 * HH * KK;
        build_w_kernel<<<(n + 255) / 256, 256, 0, stream>>>(Vi, Vf, Vo, Vg,
                                                            Ui, Uf, Uo, Ug, W);
    }
    {
        int n = BB * HH;
        init_state_kernel<<<(n + 255) / 256, 256, 0, stream>>>(h0, c0, hb0, cbuf, n);
    }

    // Sequential recurrence: 512 step kernels, h ping-pongs in bf16.
    for (int t = 0; t < TT; ++t) {
        const __bf16* hprev = (t & 1) ? hb1 : hb0;
        __bf16*       hnext = (t & 1) ? hb0 : hb1;
        lstm_step_kernel<<<HH / 32, 256, 0, stream>>>(W, xb, hprev, hnext, cbuf,
                                                      bi, bf_, bo, bg, out, t);
    }
}
